// SparseMoE_7258494730453
// MI455X (gfx1250) — matrix operand order repaired and checked by reading, not compiled, not run on hardware
//
#include <hip/hip_runtime.h>
#include <hip/hip_bf16.h>

// Sparse MoE (B=4, L=2048, D=1024, E=8, FF=4096, top_k from device) for gfx1250.
// Strategy: bf16 WMMA (v_wmma_f32_16x16x32_bf16). One-time fp32->bf16
// transpose of w1/w2 into d_ws (~129 MB needed) so WMMA B operands are
// lane-contiguous; weights then fit in the 192MB L2 and are streamed from L2.

#define T_TOK 8192
#define DDIM  1024
#define NEXP  8
#define FFDIM 4096
#define MTILE 32      // tokens per workgroup
#define FFC   256     // FF chunk staged through LDS per iteration

typedef __attribute__((ext_vector_type(16))) __bf16 bf16x16;
typedef __attribute__((ext_vector_type(8)))  float  f32x8;

union BfVec { uint4 u[2]; bf16x16 v; };

static __device__ inline f32x8 zero8() {
  f32x8 z = {0.f, 0.f, 0.f, 0.f, 0.f, 0.f, 0.f, 0.f};
  return z;
}

static __device__ inline float gelu_exact(float v) {
  return 0.5f * v * (1.0f + erff(v * 0.70710678118654752440f));
}

// ---------------------------------------------------------------------------
// Kernel 0: gating. One wave (32 lanes) per token: 8 dot products over D,
// wave32 xor-shuffle reduction, then softmax + top-k + normalize on lane 0.
// ---------------------------------------------------------------------------
__global__ void moe_gate_kernel(const float* __restrict__ x,
                                const float* __restrict__ gate_w,
                                const int* __restrict__ topk_p,
                                float* __restrict__ comb) {
  const int wave = threadIdx.x >> 5;   // 8 waves per block
  const int lane = threadIdx.x & 31;
  const int tok = blockIdx.x * 8 + wave;
  if (tok >= T_TOK) return;

  float acc[NEXP];
#pragma unroll
  for (int e = 0; e < NEXP; ++e) acc[e] = 0.f;

  const float* xr = x + (size_t)tok * DDIM;
  for (int d = lane; d < DDIM; d += 32) {
    const float xv = xr[d];
    const float* g = gate_w + (size_t)d * NEXP;
#pragma unroll
    for (int e = 0; e < NEXP; ++e) acc[e] += xv * g[e];
  }
#pragma unroll
  for (int e = 0; e < NEXP; ++e)
    for (int off = 16; off > 0; off >>= 1)
      acc[e] += __shfl_xor(acc[e], off, 32);

  if (lane == 0) {
    int k = *topk_p;
    if (k < 1) k = 1;
    if (k > NEXP) k = NEXP;
    float mx = acc[0];
    for (int e = 1; e < NEXP; ++e) mx = fmaxf(mx, acc[e]);
    float p[NEXP], sum = 0.f;
    for (int e = 0; e < NEXP; ++e) { p[e] = expf(acc[e] - mx); sum += p[e]; }
    const float inv_s = 1.f / sum;
    for (int e = 0; e < NEXP; ++e) p[e] *= inv_s;
    float outp[NEXP];
    bool used[NEXP];
    for (int e = 0; e < NEXP; ++e) { outp[e] = 0.f; used[e] = false; }
    float wsum = 0.f;
    for (int i = 0; i < k; ++i) {        // ties -> lowest index (matches top_k)
      int best = 0; float bv = -1.f;
      for (int e = 0; e < NEXP; ++e)
        if (!used[e] && p[e] > bv) { bv = p[e]; best = e; }
      used[best] = true; outp[best] = bv; wsum += bv;
    }
    const float inv_w = 1.f / wsum;
    for (int e = 0; e < NEXP; ++e)
      comb[(size_t)tok * NEXP + e] = outp[e] * inv_w;
  }
}

// ---------------------------------------------------------------------------
// Kernel 1/2: batched 32x32-tiled transpose with fp32 -> bf16 conversion.
// in: [batch][rows][cols] fp32, out: [batch][cols][rows] bf16.
// ---------------------------------------------------------------------------
__global__ void transpose_cvt_kernel(const float* __restrict__ in,
                                     __bf16* __restrict__ out,
                                     int rows, int cols) {
  __shared__ float tile[32][33];
  const int b = blockIdx.z;
  const float* ip = in + (size_t)b * rows * cols;
  __bf16* op = out + (size_t)b * rows * cols;
  const int c0 = blockIdx.x * 32;
  const int r0 = blockIdx.y * 32;
  const int tx = threadIdx.x;          // 32
  const int ty = threadIdx.y;          // 8
#pragma unroll
  for (int i = 0; i < 32; i += 8)
    tile[ty + i][tx] = ip[(size_t)(r0 + ty + i) * cols + (c0 + tx)];
  __syncthreads();
#pragma unroll
  for (int i = 0; i < 32; i += 8)
    op[(size_t)(c0 + ty + i) * rows + (r0 + tx)] = (__bf16)tile[tx][ty + i];
}

// ---------------------------------------------------------------------------
// Kernel 3: fused MoE expert compute.
// Grid: T/32 workgroups, 512 threads (16 wave32s).
//   Wave w: phase1 computes H[32 x 16] columns [w*16, w*16+16) of the FF
//   chunk; phase2 owns output columns [w*64, w*64+64) (8 accumulator tiles).
// ---------------------------------------------------------------------------
__launch_bounds__(512, 1)
__global__ void moe_expert_kernel(const float* __restrict__ x,
                                  const float* __restrict__ b1,   // [E][FF]
                                  const float* __restrict__ b2,   // [E][D]
                                  const __bf16* __restrict__ w1t, // [E][FF][D]
                                  const __bf16* __restrict__ w2t, // [E][D][FF]
                                  const float* __restrict__ comb, // [T][E]
                                  float* __restrict__ out) {
  __shared__ __align__(16) __bf16 lds_x[MTILE][DDIM + 8];  // pad: conflict-free b128
  __shared__ __align__(16) __bf16 lds_h[MTILE][FFC + 8];
  __shared__ float lds_comb[MTILE][NEXP];

  const int tid   = threadIdx.x;
  const int wave  = tid >> 5;      // 0..15
  const int lane  = tid & 31;
  const int lhalf = lane >> 4;     // 0 | 1
  const int l16   = lane & 15;
  const int t0    = blockIdx.x * MTILE;

  // Stage x tile as bf16 in LDS (read fp32 as float4, coalesced).
  for (int idx = tid; idx < MTILE * (DDIM / 4); idx += (int)blockDim.x) {
    const int row  = idx / (DDIM / 4);
    const int col4 = idx % (DDIM / 4);
    const float4 v = ((const float4*)(x + (size_t)(t0 + row) * DDIM))[col4];
    __bf16* dst = &lds_x[row][col4 * 4];
    dst[0] = (__bf16)v.x; dst[1] = (__bf16)v.y;
    dst[2] = (__bf16)v.z; dst[3] = (__bf16)v.w;
  }
  if (tid < MTILE * NEXP)
    lds_comb[tid / NEXP][tid % NEXP] = comb[(size_t)t0 * NEXP + tid];
  __syncthreads();

  // Output accumulators: [row-tile 0..1][d-tile 0..3], 16x16 f32 each.
  f32x8 outC[2][4];
#pragma unroll
  for (int rt = 0; rt < 2; ++rt)
#pragma unroll
    for (int dt = 0; dt < 4; ++dt) outC[rt][dt] = zero8();

  for (int e = 0; e < NEXP; ++e) {
    // Uniform expert-skip: true routed sparsity when a tile misses an expert.
    bool active = false;
    for (int m = 0; m < MTILE; ++m) active |= (lds_comb[m][e] != 0.f);
    if (!active) continue;

    const __bf16* w1e = w1t + (size_t)e * FFDIM * DDIM;
    const __bf16* w2e = w2t + (size_t)e * DDIM * FFDIM;

    for (int ff0 = 0; ff0 < FFDIM; ff0 += FFC) {
      // ---- Phase 1: H[32][ffc..ffc+16) = comb * gelu(x @ w1 + b1) ----
      const int ffc = wave * 16;  // this wave's column group within the chunk
      f32x8 c[2] = {zero8(), zero8()};
      // B operand: lane l16 = column N, K contiguous in memory (w1 transposed).
      const __bf16* brow =
          w1e + (size_t)(ff0 + ffc + l16) * DDIM + lhalf * 8;
      for (int k = 0; k < DDIM; k += 32) {
        BfVec bv;
        bv.u[0] = *(const uint4*)(brow + k);
        bv.u[1] = *(const uint4*)(brow + k + 16);
#pragma unroll
        for (int rt = 0; rt < 2; ++rt) {
          // A layout (16-bit 16x32): lanes0-15 K[0..7]+K[16..23],
          // lanes16-31 K[8..15]+K[24..31], M = lane&15.
          const __bf16* arow = &lds_x[rt * 16 + l16][k + lhalf * 8];
          BfVec av;
          av.u[0] = *(const uint4*)(arow);
          av.u[1] = *(const uint4*)(arow + 16);
          c[rt] = __builtin_amdgcn_wmma_f32_16x16x32_bf16(
              false, av.v, false, bv.v, (short)0, c[rt], false, false);
        }
      }
      const float b1v = b1[(size_t)e * FFDIM + ff0 + ffc + l16];
      __syncthreads();  // previous phase-2 readers of lds_h are done
#pragma unroll
      for (int rt = 0; rt < 2; ++rt) {
#pragma unroll
        for (int r = 0; r < 8; ++r) {
          const int m = rt * 16 + lhalf * 8 + r;    // C/D layout: M = r + 8*half
          const float v = c[rt][r] + b1v;
          lds_h[m][ffc + l16] = (__bf16)(gelu_exact(v) * lds_comb[m][e]);
        }
      }
      __syncthreads();  // lds_h ready for everyone

      // ---- Phase 2: outC += H @ w2 over this FF chunk ----
      for (int k = 0; k < FFC; k += 32) {
        BfVec a2[2];
#pragma unroll
        for (int rt = 0; rt < 2; ++rt) {
          const __bf16* hrow = &lds_h[rt * 16 + l16][k + lhalf * 8];
          a2[rt].u[0] = *(const uint4*)(hrow);
          a2[rt].u[1] = *(const uint4*)(hrow + 16);
        }
#pragma unroll
        for (int dt = 0; dt < 4; ++dt) {
          const __bf16* b2row =
              w2e + (size_t)(wave * 64 + dt * 16 + l16) * FFDIM +
              (ff0 + k + lhalf * 8);
          BfVec bv;
          bv.u[0] = *(const uint4*)(b2row);
          bv.u[1] = *(const uint4*)(b2row + 16);
#pragma unroll
          for (int rt = 0; rt < 2; ++rt)
            outC[rt][dt] = __builtin_amdgcn_wmma_f32_16x16x32_bf16(
                false, a2[rt].v, false, bv.v, (short)0, outC[rt][dt],
                false, false);
        }
      }
    }

    // comb-scaled b2 bias term: out += comb[t,e] * b2[e,d]
#pragma unroll
    for (int dt = 0; dt < 4; ++dt) {
      const float b2v = b2[(size_t)e * DDIM + wave * 64 + dt * 16 + l16];
#pragma unroll
      for (int rt = 0; rt < 2; ++rt)
#pragma unroll
        for (int r = 0; r < 8; ++r) {
          const int m = rt * 16 + lhalf * 8 + r;
          outC[rt][dt][r] += lds_comb[m][e] * b2v;
        }
    }
  }

  // Store fp32 output.
#pragma unroll
  for (int rt = 0; rt < 2; ++rt)
#pragma unroll
    for (int dt = 0; dt < 4; ++dt)
#pragma unroll
      for (int r = 0; r < 8; ++r) {
        const int m = rt * 16 + lhalf * 8 + r;
        out[(size_t)(t0 + m) * DDIM + wave * 64 + dt * 16 + l16] =
            outC[rt][dt][r];
      }
}

// ---------------------------------------------------------------------------
extern "C" void kernel_launch(void* const* d_in, const int* in_sizes, int n_in,
                              void* d_out, int out_size, void* d_ws,
                              size_t ws_size, hipStream_t stream) {
  const float* x      = (const float*)d_in[0];
  const float* gate_w = (const float*)d_in[1];
  const float* w1     = (const float*)d_in[2];
  const float* b1     = (const float*)d_in[3];
  const float* w2     = (const float*)d_in[4];
  const float* b2     = (const float*)d_in[5];
  const int*   topk   = (const int*)d_in[6];
  float*       out    = (float*)d_out;

  // Workspace layout (needs ~129 MB):
  //   comb : T*E fp32                  = 256 KB
  //   w1t  : E*FF*D bf16 (transposed)  = 64 MB
  //   w2t  : E*D*FF bf16 (transposed)  = 64 MB
  char* ws = (char*)d_ws;
  float*  comb = (float*)ws;
  __bf16* w1t  = (__bf16*)(ws + (size_t)T_TOK * NEXP * sizeof(float));
  __bf16* w2t  = w1t + (size_t)NEXP * FFDIM * DDIM;

  moe_gate_kernel<<<T_TOK / 8, 256, 0, stream>>>(x, gate_w, topk, comb);

  // w1[E,D,FF] -> w1t[E,FF,D]
  transpose_cvt_kernel<<<dim3(FFDIM / 32, DDIM / 32, NEXP), dim3(32, 8), 0,
                         stream>>>(w1, w1t, DDIM, FFDIM);
  // w2[E,FF,D] -> w2t[E,D,FF]
  transpose_cvt_kernel<<<dim3(DDIM / 32, FFDIM / 32, NEXP), dim3(32, 8), 0,
                         stream>>>(w2, w2t, FFDIM, DDIM);

  moe_expert_kernel<<<T_TOK / MTILE, 512, 0, stream>>>(x, b1, b2, w1t, w2t,
                                                       comb, out);
}
